// TransformerEncoderLayer_86603720557328
// MI455X (gfx1250) — matrix operand order repaired and checked by reading, not compiled, not run on hardware
//
#include <hip/hip_runtime.h>
#include <math.h>

// ---------------------------------------------------------------------------
// Types for CDNA5 WMMA (wave32)
// ---------------------------------------------------------------------------
typedef __attribute__((ext_vector_type(8)))  float          v8f;
typedef __attribute__((ext_vector_type(8)))  unsigned short v8u16;
typedef __attribute__((ext_vector_type(16))) unsigned short v16u16;
typedef __attribute__((ext_vector_type(16))) __bf16         v16bf;

union u16x16 { v8u16 h[2]; v16u16 v; unsigned short u[16]; };

static __device__ __forceinline__ unsigned short f2bf(float f) {
    unsigned int u = __float_as_uint(f);
    unsigned int r = u + 0x7FFFu + ((u >> 16) & 1u);   // round-to-nearest-even
    return (unsigned short)(r >> 16);
}

static __device__ __forceinline__ v8f wmma_bf16(v16u16 a, v16u16 b, v8f c) {
    // D(f32 16x16) = A(bf16 16x32) * B(bf16 32x16) + C
    return __builtin_amdgcn_wmma_f32_16x16x32_bf16(
        /*neg_a=*/false, __builtin_bit_cast(v16bf, a),
        /*neg_b=*/false, __builtin_bit_cast(v16bf, b),
        /*c_mod=*/(short)0, c, /*reuse_a=*/false, /*reuse_b=*/false);
}

// ---------------------------------------------------------------------------
// Problem constants
// ---------------------------------------------------------------------------
#define D_MODEL 1024
#define NHEAD   16
#define HEAD_DIM 64
#define D_FF    4096
#define SEQ     2048
#define BATCH   2
#define NROWS   (SEQ * BATCH)   // 4096 flattened (s,b) rows

// ---------------------------------------------------------------------------
// fp32 -> bf16 convert (no transpose)
// ---------------------------------------------------------------------------
__global__ void cvt_bf16_kernel(const float* __restrict__ in,
                                unsigned short* __restrict__ out, int n) {
    int i = blockIdx.x * blockDim.x + threadIdx.x;
    int stride = gridDim.x * blockDim.x;
    for (; i < n; i += stride) out[i] = f2bf(in[i]);
}

// ---------------------------------------------------------------------------
// fp32 (rows x cols) -> bf16 transposed (cols x rows).  rows, cols % 32 == 0
// ---------------------------------------------------------------------------
__global__ void transpose_cvt_bf16_kernel(const float* __restrict__ in,
                                          unsigned short* __restrict__ out,
                                          int rows, int cols) {
    __shared__ float tile[32][33];
    int x  = blockIdx.x * 32 + threadIdx.x;   // input col
    int y0 = blockIdx.y * 32;                 // input row base
    #pragma unroll
    for (int j = threadIdx.y; j < 32; j += 8)
        tile[j][threadIdx.x] = in[(size_t)(y0 + j) * cols + x];
    __syncthreads();
    int ox  = y0 + threadIdx.x;               // output col (= input row)
    int oy0 = blockIdx.x * 32;                // output row base (= input col)
    #pragma unroll
    for (int j = threadIdx.y; j < 32; j += 8)
        out[(size_t)(oy0 + j) * rows + ox] = f2bf(tile[threadIdx.x][j]);
}

// ---------------------------------------------------------------------------
// Generic bf16 WMMA GEMM:  C[M,N] = A[M,K] @ Bt[N,K]^T + bias
// Block = 128 threads (4 waves); wave tile = 16 rows x 64 cols; K step 32.
// Software-pipelined: fragments for k+32 are loaded while WMMAs for k issue.
// ---------------------------------------------------------------------------
template <bool RELU, bool OUTF, bool OUTB>
__global__ __launch_bounds__(128, 2)
void gemm_bf16_kernel(const unsigned short* __restrict__ A,
                      const unsigned short* __restrict__ Bt,
                      const float* __restrict__ bias,
                      float* __restrict__ Cf,
                      unsigned short* __restrict__ Cb,
                      int M, int N, int K) {
    const int lane = threadIdx.x & 31;
    const int wave = threadIdx.x >> 5;
    const int l15  = lane & 15;
    const int hi   = lane >> 4;
    const int rowBase = blockIdx.y * 64 + wave * 16;
    const int colBase = blockIdx.x * 64;

    v8f acc[4] = {};

    // A fragment: lane row = rowBase + l15; e -> K = kb+hi*16+e
    const unsigned short* ap  = A + (size_t)(rowBase + l15) * K + hi * 16;
    // B fragment: lane col = colBase+nt*16+l15; e -> K = kb+hi*16+e
    const unsigned short* bp0 = Bt + (size_t)(colBase + l15) * K + hi * 16;

    u16x16 a_cur, a_nxt;
    u16x16 b_cur[4], b_nxt[4];

    a_cur.h[0] = *(const v8u16*)(ap);
    a_cur.h[1] = *(const v8u16*)(ap + 8);
    #pragma unroll
    for (int nt = 0; nt < 4; ++nt)
        b_cur[nt].v = *(const v16u16*)(bp0 + (size_t)nt * 16 * K);

    #pragma unroll 2
    for (int kb = 0; kb < K; kb += 32) {
        const int kn = (kb + 32 < K) ? kb + 32 : 0;   // harmless wrap on last iter
        a_nxt.h[0] = *(const v8u16*)(ap + kn);
        a_nxt.h[1] = *(const v8u16*)(ap + kn + 8);
        #pragma unroll
        for (int nt = 0; nt < 4; ++nt)
            b_nxt[nt].v = *(const v16u16*)(bp0 + (size_t)nt * 16 * K + kn);

        #pragma unroll
        for (int nt = 0; nt < 4; ++nt)
            acc[nt] = wmma_bf16(a_cur.v, b_cur[nt].v, acc[nt]);

        a_cur = a_nxt;
        #pragma unroll
        for (int nt = 0; nt < 4; ++nt) b_cur[nt] = b_nxt[nt];
    }

    #pragma unroll
    for (int nt = 0; nt < 4; ++nt) {
        const int col = colBase + nt * 16 + l15;
        const float bv = bias[col];
        #pragma unroll
        for (int r = 0; r < 8; ++r) {
            const int row = rowBase + r + 8 * hi;   // C/D layout: M = r + 8*hi
            float v = acc[nt][r] + bv;
            if (RELU) v = fmaxf(v, 0.0f);
            const size_t idx = (size_t)row * N + col;
            if (OUTF) Cf[idx] = v;
            if (OUTB) Cb[idx] = f2bf(v);
        }
    }
}

// ---------------------------------------------------------------------------
// RoPE + head packing.
// q,k,v: fp32 [NROWS, D_MODEL], row = s*BATCH + b.
// qh,kh: bf16 [b*NHEAD+h][s][HEAD_DIM];  vt: bf16 [b*NHEAD+h][HEAD_DIM][s]
// ---------------------------------------------------------------------------
__global__ void rope_pack_kernel(const float* __restrict__ q,
                                 const float* __restrict__ k,
                                 const float* __restrict__ v,
                                 unsigned short* __restrict__ qh,
                                 unsigned short* __restrict__ kh,
                                 unsigned short* __restrict__ vt) {
    const int d  = threadIdx.x;        // 0..63
    const int s  = blockIdx.x;         // 0..SEQ-1
    const int bh = blockIdx.y;         // 0..31
    const int b  = bh >> 4;
    const int h  = bh & 15;

    const size_t in_idx = ((size_t)s * BATCH + b) * D_MODEL + h * HEAD_DIM + d;
    const size_t pr_idx = ((size_t)s * BATCH + b) * D_MODEL + h * HEAD_DIM + (d ^ 1);

    const int   i   = d >> 1;
    const float inv = __powf(10000.0f, -(float)(2 * i) / (float)HEAD_DIM);
    const float ang = (float)s * inv;
    const float c = cosf(ang), sn = sinf(ang);

    const float qv = q[in_idx], kv = k[in_idx];
    const float qp = q[pr_idx], kp = k[pr_idx];
    const float qrot = (d & 1) ? qp : -qp;   // rotate_every_two
    const float krot = (d & 1) ? kp : -kp;

    const size_t hidx = ((size_t)bh * SEQ + s) * HEAD_DIM + d;
    qh[hidx] = f2bf(qv * c + qrot * sn);
    kh[hidx] = f2bf(kv * c + krot * sn);
    vt[((size_t)bh * HEAD_DIM + d) * SEQ + s] = f2bf(v[in_idx]);
}

// ---------------------------------------------------------------------------
// Flash attention, one wave per (b,h, 16-query block).
// S = (Q Kt)/8 with mask; online softmax; O = P V.  Output bf16 [row, D_MODEL].
// P tile bounced through LDS as bf16 so the A-fragment reload is two
// ds_load_b128 per lane.  __launch_bounds__(32,1) releases the full VGPR file
// to the single wave so Q fragments / accumulators never spill to scratch.
// ---------------------------------------------------------------------------
__global__ __launch_bounds__(32, 1)
void attention_kernel(const unsigned short* __restrict__ qh,
                      const unsigned short* __restrict__ kh,
                      const unsigned short* __restrict__ vt,
                      const float* __restrict__ mask,
                      unsigned short* __restrict__ out) {
    const int lane = threadIdx.x & 31;
    const int l15  = lane & 15;
    const int hi   = lane >> 4;
    const int q0   = blockIdx.x * 16;
    const int bh   = blockIdx.y;
    const int b    = bh >> 4;
    const int h    = bh & 15;

    __shared__ unsigned short pbuf[16 * 32];   // P tile, row-major 16x32 bf16

    const unsigned short* qbase = qh + (size_t)bh * SEQ * HEAD_DIM;
    const unsigned short* kbase = kh + (size_t)bh * SEQ * HEAD_DIM;
    const unsigned short* vbase = vt + (size_t)bh * HEAD_DIM * SEQ;

    // Q A-fragments: rows q0..q0+15, d split 0..31 / 32..63
    const unsigned short* qp = qbase + (size_t)(q0 + l15) * HEAD_DIM + hi * 16;
    u16x16 qa0, qa1;
    qa0.h[0] = *(const v8u16*)(qp);
    qa0.h[1] = *(const v8u16*)(qp + 8);
    qa1.h[0] = *(const v8u16*)(qp + 32);
    qa1.h[1] = *(const v8u16*)(qp + 40);

    v8f acc[4] = {};
    float mrun[8], lrun[8];
    #pragma unroll
    for (int r = 0; r < 8; ++r) { mrun[r] = -1e30f; lrun[r] = 0.0f; }

    for (int kb = 0; kb < SEQ; kb += 32) {
        // Issue V B-fragment loads first so they overlap S-WMMAs + softmax.
        u16x16 vb[4];
        #pragma unroll
        for (int nt = 0; nt < 4; ++nt) {
            const unsigned short* vp =
                vbase + (size_t)(nt * 16 + l15) * SEQ + kb + hi * 16;
            vb[nt].v = *(const v16u16*)vp;
        }

        float st[2][8];
        #pragma unroll
        for (int t = 0; t < 2; ++t) {
            const int key = kb + t * 16 + l15;
            const unsigned short* kp = kbase + (size_t)key * HEAD_DIM + hi * 16;
            u16x16 b0, b1;
            b0.v = *(const v16u16*)kp;          // d = hi*16 + e  (0..31)
            b1.v = *(const v16u16*)(kp + 32);   // d = 32 + hi*16 + e
            v8f c = {};
            c = wmma_bf16(qa0.v, b0.v, c);
            c = wmma_bf16(qa1.v, b1.v, c);
            #pragma unroll
            for (int r = 0; r < 8; ++r) {
                const int qi = q0 + r + 8 * hi;
                const float mv = mask[(size_t)qi * SEQ + key];
                const float sv = c[r] * 0.125f;   // 1/sqrt(64)
                st[t][r] = (mv < 0.1f) ? -__builtin_inff() : sv;
            }
        }

        // Online softmax (row stats live in C-layout: row = r + 8*hi)
        #pragma unroll
        for (int r = 0; r < 8; ++r) {
            float tm = fmaxf(st[0][r], st[1][r]);
            tm = fmaxf(tm, __shfl_xor(tm, 1, 32));
            tm = fmaxf(tm, __shfl_xor(tm, 2, 32));
            tm = fmaxf(tm, __shfl_xor(tm, 4, 32));
            tm = fmaxf(tm, __shfl_xor(tm, 8, 32));
            const float mnew = fmaxf(mrun[r], tm);
            float alpha, p0, p1;
            if (mnew > -1e29f) {
                alpha = __expf(mrun[r] - mnew);
                p0 = __expf(st[0][r] - mnew);
                p1 = __expf(st[1][r] - mnew);
                mrun[r] = mnew;
            } else {
                alpha = 1.0f; p0 = 0.0f; p1 = 0.0f;
            }
            float rs = p0 + p1;
            rs += __shfl_xor(rs, 1, 32);
            rs += __shfl_xor(rs, 2, 32);
            rs += __shfl_xor(rs, 4, 32);
            rs += __shfl_xor(rs, 8, 32);
            lrun[r] = lrun[r] * alpha + rs;
            #pragma unroll
            for (int nt = 0; nt < 4; ++nt) acc[nt][r] *= alpha;
            pbuf[(r + 8 * hi) * 32 + l15]      = f2bf(p0);
            pbuf[(r + 8 * hi) * 32 + 16 + l15] = f2bf(p1);
        }
        __syncthreads();

        // Re-load P as bf16 A-fragment: row = l15, kk = hi*16 + e.
        // Both halves are 8 consecutive ushorts -> 16B-aligned LDS vector loads.
        u16x16 pa;
        pa.h[0] = *(const v8u16*)&pbuf[l15 * 32 + hi * 16];
        pa.h[1] = *(const v8u16*)&pbuf[l15 * 32 + hi * 16 + 8];
        __syncthreads();

        // O += P @ V   (B frag from V^T: col d = nt*16+l15, K = kb+hi*16+e)
        #pragma unroll
        for (int nt = 0; nt < 4; ++nt)
            acc[nt] = wmma_bf16(pa.v, vb[nt].v, acc[nt]);
    }

    // Normalize and store bf16 at [(qi*BATCH + b), h*64 + d]
    #pragma unroll
    for (int nt = 0; nt < 4; ++nt) {
        #pragma unroll
        for (int r = 0; r < 8; ++r) {
            const int qi = q0 + r + 8 * hi;
            const float l = lrun[r];
            const float val = (l > 0.0f) ? acc[nt][r] / l : 0.0f;
            out[((size_t)qi * BATCH + b) * D_MODEL + h * HEAD_DIM + nt * 16 + l15] =
                f2bf(val);
        }
    }
}

// ---------------------------------------------------------------------------
// LayerNorm with fused residual:  y = LN(x + resid) * g + beta
// One row (1024 cols) per block of 256 threads.
// ---------------------------------------------------------------------------
__global__ void layernorm_kernel(const float* __restrict__ x,
                                 const float* __restrict__ resid,
                                 const float* __restrict__ g,
                                 const float* __restrict__ beta,
                                 float* __restrict__ outf,
                                 unsigned short* __restrict__ outb) {
    __shared__ float red[256];
    const int row = blockIdx.x;
    const int tid = threadIdx.x;
    const float* xr = x + (size_t)row * D_MODEL;
    const float* rr = resid + (size_t)row * D_MODEL;

    float vals[4];
    float s = 0.0f;
    #pragma unroll
    for (int i = 0; i < 4; ++i) {
        const int c = tid + i * 256;
        vals[i] = xr[c] + rr[c];
        s += vals[i];
    }
    red[tid] = s;
    __syncthreads();
    for (int o = 128; o > 0; o >>= 1) {
        if (tid < o) red[tid] += red[tid + o];
        __syncthreads();
    }
    const float mu = red[0] * (1.0f / D_MODEL);
    __syncthreads();

    float s2 = 0.0f;
    #pragma unroll
    for (int i = 0; i < 4; ++i) {
        const float d = vals[i] - mu;
        s2 += d * d;
    }
    red[tid] = s2;
    __syncthreads();
    for (int o = 128; o > 0; o >>= 1) {
        if (tid < o) red[tid] += red[tid + o];
        __syncthreads();
    }
    const float var = red[0] * (1.0f / D_MODEL);
    const float inv = rsqrtf(var + 1e-5f);

    #pragma unroll
    for (int i = 0; i < 4; ++i) {
        const int c = tid + i * 256;
        const float y = (vals[i] - mu) * inv * g[c] + beta[c];
        if (outf) outf[(size_t)row * D_MODEL + c] = y;
        if (outb) outb[(size_t)row * D_MODEL + c] = f2bf(y);
    }
}

// ---------------------------------------------------------------------------
// Host-side orchestration
// ---------------------------------------------------------------------------
extern "C" void kernel_launch(void* const* d_in, const int* in_sizes, int n_in,
                              void* d_out, int out_size, void* d_ws, size_t ws_size,
                              hipStream_t stream) {
    const float* src      = (const float*)d_in[0];
    const float* src_mask = (const float*)d_in[1];
    const float* Wq = (const float*)d_in[2];  const float* bq = (const float*)d_in[3];
    const float* Wk = (const float*)d_in[4];  const float* bk = (const float*)d_in[5];
    const float* Wv = (const float*)d_in[6];  const float* bv = (const float*)d_in[7];
    const float* Wo = (const float*)d_in[8];  const float* bo = (const float*)d_in[9];
    const float* W1 = (const float*)d_in[10]; const float* b1 = (const float*)d_in[11];
    const float* W2 = (const float*)d_in[12]; const float* b2 = (const float*)d_in[13];
    const float* g1 = (const float*)d_in[14]; const float* be1 = (const float*)d_in[15];
    const float* g2 = (const float*)d_in[16]; const float* be2 = (const float*)d_in[17];
    float* out = (float*)d_out;

    // ---- workspace carve-up (bytes, 256-aligned) ----
    char* w = (char*)d_ws;
    size_t off = 0;
    auto alloc = [&](size_t bytes) {
        char* p = w + off;
        off += (bytes + 255) & ~(size_t)255;
        return p;
    };
    const size_t BF = sizeof(unsigned short);
    unsigned short* WqT  = (unsigned short*)alloc((size_t)D_MODEL * D_MODEL * BF);
    unsigned short* WkT  = (unsigned short*)alloc((size_t)D_MODEL * D_MODEL * BF);
    unsigned short* WvT  = (unsigned short*)alloc((size_t)D_MODEL * D_MODEL * BF);
    unsigned short* WoT  = (unsigned short*)alloc((size_t)D_MODEL * D_MODEL * BF);
    unsigned short* W1T  = (unsigned short*)alloc((size_t)D_FF * D_MODEL * BF);
    unsigned short* W2T  = (unsigned short*)alloc((size_t)D_MODEL * D_FF * BF);
    unsigned short* srcb = (unsigned short*)alloc((size_t)NROWS * D_MODEL * BF);
    float* qf = (float*)alloc((size_t)NROWS * D_MODEL * 4);
    float* kf = (float*)alloc((size_t)NROWS * D_MODEL * 4);
    float* vf = (float*)alloc((size_t)NROWS * D_MODEL * 4);
    unsigned short* qhd = (unsigned short*)alloc((size_t)BATCH * NHEAD * SEQ * HEAD_DIM * BF);
    unsigned short* khd = (unsigned short*)alloc((size_t)BATCH * NHEAD * SEQ * HEAD_DIM * BF);
    unsigned short* vtd = (unsigned short*)alloc((size_t)BATCH * NHEAD * HEAD_DIM * SEQ * BF);
    unsigned short* attnb = (unsigned short*)alloc((size_t)NROWS * D_MODEL * BF);
    unsigned short* src2b = (unsigned short*)alloc((size_t)NROWS * D_MODEL * BF);
    unsigned short* ff1b  = (unsigned short*)alloc((size_t)NROWS * D_FF * BF);
    // alias: qf/kf are dead after rope_pack
    float* tmpf  = qf;   // attn@Wo output, later ff2 output
    float* src2f = kf;   // LN1 fp32 output (needed for final residual)

    const dim3 tb32(32, 8);

    // 1) weights -> transposed bf16
    transpose_cvt_bf16_kernel<<<dim3(D_MODEL / 32, D_MODEL / 32), tb32, 0, stream>>>(Wq, WqT, D_MODEL, D_MODEL);
    transpose_cvt_bf16_kernel<<<dim3(D_MODEL / 32, D_MODEL / 32), tb32, 0, stream>>>(Wk, WkT, D_MODEL, D_MODEL);
    transpose_cvt_bf16_kernel<<<dim3(D_MODEL / 32, D_MODEL / 32), tb32, 0, stream>>>(Wv, WvT, D_MODEL, D_MODEL);
    transpose_cvt_bf16_kernel<<<dim3(D_MODEL / 32, D_MODEL / 32), tb32, 0, stream>>>(Wo, WoT, D_MODEL, D_MODEL);
    transpose_cvt_bf16_kernel<<<dim3(D_FF / 32, D_MODEL / 32), tb32, 0, stream>>>(W1, W1T, D_MODEL, D_FF);
    transpose_cvt_bf16_kernel<<<dim3(D_MODEL / 32, D_FF / 32), tb32, 0, stream>>>(W2, W2T, D_FF, D_MODEL);

    // 2) src -> bf16
    cvt_bf16_kernel<<<1024, 256, 0, stream>>>(src, srcb, NROWS * D_MODEL);

    // 3) Q/K/V projections (fp32 out, bias fused)
    const dim3 gemm_blk(128);
    const dim3 gD(D_MODEL / 64, NROWS / 64);
    gemm_bf16_kernel<false, true, false><<<gD, gemm_blk, 0, stream>>>(srcb, WqT, bq, qf, nullptr, NROWS, D_MODEL, D_MODEL);
    gemm_bf16_kernel<false, true, false><<<gD, gemm_blk, 0, stream>>>(srcb, WkT, bk, kf, nullptr, NROWS, D_MODEL, D_MODEL);
    gemm_bf16_kernel<false, true, false><<<gD, gemm_blk, 0, stream>>>(srcb, WvT, bv, vf, nullptr, NROWS, D_MODEL, D_MODEL);

    // 4) RoPE + head packing (bf16; V transposed)
    rope_pack_kernel<<<dim3(SEQ, BATCH * NHEAD), 64, 0, stream>>>(qf, kf, vf, qhd, khd, vtd);

    // 5) Flash attention -> bf16 [row, D_MODEL]
    attention_kernel<<<dim3(SEQ / 16, BATCH * NHEAD), 32, 0, stream>>>(qhd, khd, vtd, src_mask, attnb);

    // 6) output projection (fp32)
    gemm_bf16_kernel<false, true, false><<<gD, gemm_blk, 0, stream>>>(attnb, WoT, bo, tmpf, nullptr, NROWS, D_MODEL, D_MODEL);

    // 7) LN1: src2 = LN(src + attn_proj)
    layernorm_kernel<<<NROWS, 256, 0, stream>>>(tmpf, src, g1, be1, src2f, src2b);

    // 8) FFN up + ReLU (bf16)
    gemm_bf16_kernel<true, false, true><<<dim3(D_FF / 64, NROWS / 64), gemm_blk, 0, stream>>>(src2b, W1T, b1, nullptr, ff1b, NROWS, D_FF, D_MODEL);

    // 9) FFN down (fp32)
    gemm_bf16_kernel<false, true, false><<<gD, gemm_blk, 0, stream>>>(ff1b, W2T, b2, tmpf, nullptr, NROWS, D_MODEL, D_FF);

    // 10) LN2 -> d_out
    layernorm_kernel<<<NROWS, 256, 0, stream>>>(tmpf, src2f, g2, be2, out, nullptr);

    (void)in_sizes; (void)n_in; (void)out_size; (void)ws_size;
}
